// ArgumentGCN_32_9680856285901
// MI455X (gfx1250) — matrix operand order repaired and checked by reading, not compiled, not run on hardware
//
#include <hip/hip_runtime.h>
#include <hip/hip_bf16.h>

#define Bb 16
#define Nn 512
#define Dd 256
#define Kk 12

typedef __attribute__((ext_vector_type(16))) __bf16 v16bf;
typedef __attribute__((ext_vector_type(8)))  float  v8f;

union frag_u { v16bf v; uint4 u[2]; };
union pack_u { uint4 u; __bf16 h[8]; };

// Load a 16-element bf16 fragment stored contiguously (32B) for this lane.
// Lowers to 2x b128 loads (global or ds depending on address space).
__device__ inline v16bf load_frag(const __bf16* p) {
  frag_u r;
  const uint4* q = (const uint4*)p;
  r.u[0] = q[0];
  r.u[1] = q[1];
  return r.v;
}

// Fragment-order layouts (contiguous k order, wave32):
//  A frag slot i of lane (hi,lo): row=lo, kk=hi*16+i
//  B frag slot i of lane (hi,lo): kk=hi*16+i, col=lo
// node_frag : [b][mb=8][dc=8][mt=4][lane=32][i=16]           (A frags)
// w_frag    : [k=13][dc=8][et=16][lane=32][i=16]             (B frags, k=12 -> Ws)
// trans_frag: [k=12][b][nc=16][et=16][lane=32][i=16]         (B frags)

// ---------------------------------------------------------------------------
// Prep: convert Wk (12) and Ws (1) f32 -> bf16 in B-fragment layout.
// ---------------------------------------------------------------------------
__global__ __launch_bounds__(256) void gcn_prep_w(
    const float* __restrict__ Wk, const float* __restrict__ Ws,
    __bf16* __restrict__ w_frag) {
  const int dc = blockIdx.x, k = blockIdx.y;       // dc 0..7, k 0..12
  const int t = threadIdx.x;
  const int et = t >> 4, lp = t & 15;
  const float* src = (k < Kk) ? (Wk + (size_t)k * Dd * Dd) : Ws;
  __bf16* dstb = w_frag + (((size_t)k * 8 + dc) * 16 + et) * (32 * 16);
#pragma unroll
  for (int s = 0; s < 2; ++s) {
    const int l = lp * 2 + s;
    const int lo = l & 15, hi = l >> 4;
    const int e = et * 16 + lo;
    const float* sp = src + (size_t)e * Dd + dc * 32 + hi * 16;
    pack_u p0, p1;
#pragma unroll
    for (int j = 0; j < 8; ++j) { p0.h[j] = (__bf16)sp[j]; p1.h[j] = (__bf16)sp[8 + j]; }
    uint4* d = (uint4*)(dstb + l * 16);
    d[0] = p0.u;
    d[1] = p1.u;
  }
}

// ---------------------------------------------------------------------------
// w = sigmoid(node . Ww + bw); cast node rows to bf16 A-fragment layout.
// ---------------------------------------------------------------------------
__global__ __launch_bounds__(256) void gcn_w_and_cast(
    const float* __restrict__ node, const float* __restrict__ Ww,
    const float* __restrict__ bw, float* __restrict__ w_out,
    float* __restrict__ w_ws, __bf16* __restrict__ node_frag) {
  const int lane = threadIdx.x & 31, wave = threadIdx.x >> 5;
  const int row = blockIdx.x * 8 + wave;           // = b*Nn + m
  const int b = row >> 9, m = row & (Nn - 1);
  const float* src = node + (size_t)row * Dd + lane * 8;
  float acc = 0.f;
  pack_u pk;
#pragma unroll
  for (int j = 0; j < 8; ++j) {
    const float v = src[j];
    acc += v * Ww[lane * 8 + j];
    pk.h[j] = (__bf16)v;
  }
  // d = lane*8+j  ->  dc=lane>>2, hi=(lane>>1)&1, i0=(lane&1)?8:0
  const int mb = m >> 6, mt = (m >> 4) & 3, lo = m & 15;
  const int dc = lane >> 2, hi = (lane >> 1) & 1, i0 = (lane & 1) ? 8 : 0;
  __bf16* dst = node_frag +
      (((((size_t)b * 8 + mb) * 8 + dc) * 4 + mt) * 32 + hi * 16 + lo) * 16 + i0;
  *(uint4*)dst = pk.u;
#pragma unroll
  for (int off = 16; off > 0; off >>= 1) acc += __shfl_xor(acc, off, 32);
  if (lane == 0) {
    const float ww = 1.f / (1.f + __expf(-(acc + bw[0])));
    w_out[row] = ww;   // all_weight output [B,1,N]
    w_ws[row]  = ww;
  }
}

// ---------------------------------------------------------------------------
// Fused GEMMs: k<12 : trans[k,b] = node @ Wk[k]^T  -> trans_frag (bf16)
//              k==12: self_info  = node @ Ws^T+bs  -> f32 plain
// No LDS/barriers; A/B fragments stream from global with register rotation
// so each B load is issued one WMMA (each A load 8 WMMAs) ahead of its use.
// ---------------------------------------------------------------------------
__global__ __launch_bounds__(256) void gcn_mm(
    const __bf16* __restrict__ node_frag, const __bf16* __restrict__ w_frag,
    const float* __restrict__ bs, __bf16* __restrict__ trans_frag,
    float* __restrict__ self_info) {
  const int mb = blockIdx.x, b = blockIdx.y, k = blockIdx.z;
  const int t = threadIdx.x, lane = t & 31, wave = t >> 5;
  const int mt = wave & 3, eh = wave >> 2;
  const int lo = lane & 15, hi = lane >> 4;
  v8f acc[8];
#pragma unroll
  for (int i = 0; i < 8; ++i)
#pragma unroll
    for (int j = 0; j < 8; ++j) acc[i][j] = 0.f;

  const __bf16* abase = node_frag + (((size_t)b * 8 + mb) * 8) * (4 * 32 * 16);
  const __bf16* bbase = w_frag + ((size_t)k * 8) * (16 * 32 * 16);
  const __bf16* aF = abase + (mt * 32 + lane) * 16;

  v16bf a_cur = load_frag(aF);                                       // dc=0
  v16bf b_cur = load_frag(bbase + ((size_t)(eh * 8) * 32 + lane) * 16); // dc0,et0
#pragma unroll
  for (int dc = 0; dc < 8; ++dc) {
    v16bf a_next;
    if (dc < 7) a_next = load_frag(aF + (size_t)(dc + 1) * (4 * 32 * 16));
#pragma unroll
    for (int et = 0; et < 8; ++et) {
      const int ndc = (et == 7) ? dc + 1 : dc;
      const int net = (et + 1) & 7;
      v16bf b_next;
      if (ndc < 8)
        b_next = load_frag(bbase + (size_t)ndc * (16 * 32 * 16) +
                           ((size_t)(eh * 8 + net) * 32 + lane) * 16);
      acc[et] = __builtin_amdgcn_wmma_f32_16x16x32_bf16(
          false, a_cur, false, b_cur, (short)0, acc[et], false, false);
      if (ndc < 8) b_cur = b_next;
    }
    if (dc < 7) a_cur = a_next;
  }

  if (k < Kk) {
    // D tile -> consumer B-fragment layout: nc=n>>5, lane'=((n>>4)&1)*16+lo,
    // i=n&15 = hi*8+r (contiguous over r) -> one uint4 store per tile.
    const int nc = mb * 2 + (mt >> 1);
    __bf16* ob = trans_frag + ((((size_t)k * Bb + b) * 16 + nc) * 16) * (32 * 16);
#pragma unroll
    for (int et = 0; et < 8; ++et) {
      const int etg = eh * 8 + et;
      pack_u pk;
#pragma unroll
      for (int r2 = 0; r2 < 8; ++r2) pk.h[r2] = (__bf16)acc[et][r2];
      *(uint4*)(ob + ((size_t)etg * 32 + (mt & 1) * 16 + lo) * 16 + hi * 8) = pk.u;
    }
  } else {
    float* sp = self_info + ((size_t)b * Nn + mb * 64) * Dd;
#pragma unroll
    for (int et = 0; et < 8; ++et) {
      const int e = (eh * 8 + et) * 16 + lo;
      const float bias = bs[e];
#pragma unroll
      for (int r2 = 0; r2 < 8; ++r2) {
        const int m = mt * 16 + hi * 8 + r2;
        sp[(size_t)m * Dd + e] = acc[et][r2] + bias;
      }
    }
  }
}

// ---------------------------------------------------------------------------
// agg: agg[m,e] = sum_{k,n} wg[k,m,n]*trans[k,n,e];  neigh fused from the
// single graphs stream; out = relu(self + agg/neigh).
// Pipelined: next-step graphs loads issued before the WMMA section;
// wg tile double-buffered in LDS -> one barrier per step; B frags rotated.
// Loop order nc-outer/k-inner so per-n {mask, mask*w} float2s stay in regs.
// ---------------------------------------------------------------------------
__global__ __launch_bounds__(256) void gcn_agg(
    const int* __restrict__ graphs, const int* __restrict__ node_mask,
    const float* __restrict__ w_ws, const __bf16* __restrict__ trans_frag,
    const float* __restrict__ self_info, float* __restrict__ out) {
  __shared__ __align__(16) __bf16 Asf[2][4 * 32 * 16];   // double-buffered wg tile
  __shared__ __align__(16) float2 mw_all[Nn];            // {mask_n, mask_n*w_n}
  __shared__ float maskm[64];
  __shared__ float neighAcc[64];

  const int b = blockIdx.y, mb = blockIdx.x, m0 = mb * 64;
  const int t = threadIdx.x, lane = t & 31, wave = t >> 5;
  const int mt = wave & 3, eh = wave >> 2;

  for (int i = t; i < Nn; i += 256) {
    const float mn = (float)node_mask[b * Nn + i];
    mw_all[i] = make_float2(mn, mn * w_ws[b * Nn + i]);
  }
  if (t < 64) {
    maskm[t] = (float)node_mask[b * Nn + m0 + t];
    neighAcc[t] = 0.f;
  }
  __syncthreads();

  // staging geometry: thread -> row r (0..63), cols c0..c0+7 -> one uint4
  const int r = t >> 2, c0 = (t & 3) * 8;
  const int lo_r = r & 15, mt_r = r >> 4;
  const int hi_c = c0 >> 4, i0_c = (c0 & 8) ? 8 : 0;
  const int dstoff = (mt_r * 32 + hi_c * 16 + lo_r) * 16 + i0_c;
  const float mm = maskm[r];
  const int m_abs = m0 + r;
  const int* gbase = graphs + ((size_t)b * Nn + m0 + r) * Nn + c0;
  const size_t kstride = (size_t)Bb * Nn * Nn;

  v8f acc[8];
#pragma unroll
  for (int i = 0; i < 8; ++i)
#pragma unroll
    for (int j = 0; j < 8; ++j) acc[i][j] = 0.f;
  float neigh_part = 0.f;

  float2 mwreg[8];
  {
    const float4* mp = (const float4*)&mw_all[c0];     // nc = 0
#pragma unroll
    for (int j = 0; j < 4; ++j) ((float4*)mwreg)[j] = mp[j];
  }

  auto stage = [&](int bufIdx, const int4& ga4, const int4& gb4, int nc) {
    const int gg[8] = {ga4.x, ga4.y, ga4.z, ga4.w, gb4.x, gb4.y, gb4.z, gb4.w};
    pack_u pk;
#pragma unroll
    for (int j = 0; j < 8; ++j) {
      const int n = nc * 32 + c0 + j;
      float base = (float)gg[j] * mm;       // graph * mask_m
      if (n == m_abs) base = 0.f;           // (1 - eye)
      neigh_part += base * mwreg[j].x;      // * mask_n          -> neigh
      pk.h[j] = (__bf16)(base * mwreg[j].y);// * mask_n * w_n    -> wg
    }
    *(uint4*)(Asf[bufIdx] + dstoff) = pk.u;
  };

  // prologue: stage step 0 (nc=0, k=0) into buffer 0
  int4 gA, gB;
  {
    const int4* gp = (const int4*)gbase;
    gA = gp[0]; gB = gp[1];
  }
  stage(0, gA, gB, 0);
  __syncthreads();

  int k_cur = 0, nc_cur = 0;
  for (int s = 0; s < Kk * 16; ++s) {
    const int buf = s & 1;
    int k_n = k_cur + 1, nc_n = nc_cur;
    if (k_n == Kk) { k_n = 0; nc_n = nc_cur + 1; }
    const bool have_next = (s + 1) < Kk * 16;

    // (1) prefetch next step's graphs -> in flight during WMMA section
    if (have_next) {
      const int4* gp = (const int4*)(gbase + (size_t)k_n * kstride + nc_n * 32);
      gA = gp[0]; gB = gp[1];
    }

    // (2) compute step s: A frag from LDS, rotated B frags from global
    const __bf16* tb =
        trans_frag + ((((size_t)k_cur * Bb + b) * 16 + nc_cur) * 16) * (32 * 16);
    const v16bf a = load_frag(Asf[buf] + (mt * 32 + lane) * 16);
    v16bf b_cur = load_frag(tb + ((size_t)(eh * 8) * 32 + lane) * 16);
#pragma unroll
    for (int et = 0; et < 8; ++et) {
      v16bf b_next;
      if (et < 7)
        b_next = load_frag(tb + ((size_t)(eh * 8 + et + 1) * 32 + lane) * 16);
      acc[et] = __builtin_amdgcn_wmma_f32_16x16x32_bf16(
          false, a, false, b_cur, (short)0, acc[et], false, false);
      if (et < 7) b_cur = b_next;
    }

    // (3) stage step s+1 into the other buffer
    if (have_next) {
      if (k_n == 0) {   // nc changed: reload per-n mask/weight pairs
        const float4* mp = (const float4*)&mw_all[nc_n * 32 + c0];
#pragma unroll
        for (int j = 0; j < 4; ++j) ((float4*)mwreg)[j] = mp[j];
      }
      stage(buf ^ 1, gA, gB, nc_n);
    }
    __syncthreads();
    k_cur = k_n; nc_cur = nc_n;
  }

  // neigh: 4 partials per row -> LDS ds_add_f32
  atomicAdd(&neighAcc[r], neigh_part);
  __syncthreads();
  if (t < 64) {
    const float nv = neighAcc[t];
    neighAcc[t] = 1.f / (nv >= 1.f ? nv : 1.f);
  }
  __syncthreads();

  const int lo = lane & 15, hi = lane >> 4;
  const float* sp = self_info + ((size_t)b * Nn + m0) * Dd;
  float* op = out + ((size_t)b * Nn + m0) * Dd;
#pragma unroll
  for (int et = 0; et < 8; ++et) {
    const int e = (eh * 8 + et) * 16 + lo;
#pragma unroll
    for (int rr = 0; rr < 8; ++rr) {
      const int m = mt * 16 + hi * 8 + rr;
      const float v = acc[et][rr] * neighAcc[m] + sp[(size_t)m * Dd + e];
      op[(size_t)m * Dd + e] = fmaxf(v, 0.f);
    }
  }
}

// ---------------------------------------------------------------------------
extern "C" void kernel_launch(void* const* d_in, const int* in_sizes, int n_in,
                              void* d_out, int out_size, void* d_ws, size_t ws_size,
                              hipStream_t stream) {
  const float* node      = (const float*)d_in[0];
  const float* Ww        = (const float*)d_in[1];
  const float* bw        = (const float*)d_in[2];
  const float* Ws        = (const float*)d_in[3];
  const float* bs        = (const float*)d_in[4];
  const float* Wk        = (const float*)d_in[5];
  const int*   node_mask = (const int*)d_in[6];
  const int*   graphs    = (const int*)d_in[7];

  float* out   = (float*)d_out;
  float* w_out = out + (size_t)Bb * Nn * Dd;       // all_weight tail [B,1,N]

  char* ws = (char*)d_ws;
  const size_t node_bytes  = (size_t)Bb * Nn * Dd * 2;              // 4 MB
  const size_t trans_bytes = (size_t)Kk * Bb * Nn * Dd * 2;         // 48 MB
  const size_t wfrag_bytes = (size_t)(Kk + 1) * Dd * Dd * 2;        // 1.7 MB
  const size_t self_bytes  = (size_t)Bb * Nn * Dd * 4;              // 8 MB
  __bf16* node_frag  = (__bf16*)ws;
  __bf16* trans_frag = (__bf16*)(ws + node_bytes);
  __bf16* w_frag     = (__bf16*)(ws + node_bytes + trans_bytes);
  float*  self_info  = (float*)(ws + node_bytes + trans_bytes + wfrag_bytes);
  float*  w_ws       = (float*)(ws + node_bytes + trans_bytes + wfrag_bytes + self_bytes);

  gcn_prep_w    <<<dim3(8, Kk + 1), 256, 0, stream>>>(Wk, Ws, w_frag);
  gcn_w_and_cast<<<Bb * Nn / 8, 256, 0, stream>>>(node, Ww, bw, w_out, w_ws, node_frag);
  gcn_mm        <<<dim3(8, Bb, Kk + 1), 256, 0, stream>>>(node_frag, w_frag, bs,
                                                          trans_frag, self_info);
  gcn_agg       <<<dim3(8, Bb), 256, 0, stream>>>(graphs, node_mask, w_ws, trans_frag,
                                                  self_info, out);
}
